// ResidualAttentionBlock_Ours_86620900426085
// MI455X (gfx1250) — matrix operand order repaired and checked by reading, not compiled, not run on hardware
//
#include <hip/hip_runtime.h>
#include <hip/hip_bf16.h>
#include <cstdint>

// ---------------- constants (from reference) ----------------
#define NTOK   577
#define NB     16
#define DIM    1024
#define NH     16
#define HD     64
#define MTOK   (NTOK*NB)        // 9232 tokens
#define LEFT   519              // ceil(0.9*576)
#define K1     403              // left - 2*merged
#define NCLUST 116              // left - k1
#define NCOMP  57               // 576 - left
#define NOUT   409              // 1 + 403 + 4 + 1
#define MOUT   (NOUT*NB)        // 6544
#define FF     4096
#define LDSW   608              // 19*32 padded score width

typedef __attribute__((ext_vector_type(16))) _Float16 v16h;
typedef __attribute__((ext_vector_type(8)))  float    v8f;

// ---------------- CDNA5 async-to-LDS primitives ----------------
#if defined(__has_builtin)
#if __has_builtin(__builtin_amdgcn_s_wait_asynccnt)
#define WAIT_ASYNC(n) __builtin_amdgcn_s_wait_asynccnt(n)
#endif
#endif
#ifndef WAIT_ASYNC
#define WAIT_ASYNC(n) asm volatile("s_wait_asynccnt %0" :: "i"(n) : "memory")
#endif

// GLOBAL_LOAD_ASYNC_TO_LDS_B128: per-lane LDS[vdst] = MEM[vaddr], ASYNCcnt-tracked
__device__ __forceinline__ void async_ld_b128(_Float16* lds_dst, const _Float16* gsrc) {
  uint32_t loff = (uint32_t)(uintptr_t)lds_dst;   // LDS aperture: addr[31:0] = LDS offset
  asm volatile("global_load_async_to_lds_b128 %0, %1, off"
               :: "v"(loff), "v"(gsrc) : "memory");
}

// ---------------- WMMA helpers ----------------
__device__ __forceinline__ v8f wmma_f16(v16h a, v16h b, v8f c) {
  return __builtin_amdgcn_wmma_f32_16x16x32_f16(false, a, false, b, (short)0, c, false, false);
}

union AFragU { v16h v; float4 f[2]; };

// A tile 16x32 from row-major f16 [.., ld]; lane&15 = row, K split per ISA layout
__device__ __forceinline__ v16h load_a_frag(const _Float16* __restrict__ A, int ld,
                                            int r0, int k0, int maxRow, int lane) {
  int row = r0 + (lane & 15);
  if (row > maxRow) row = maxRow;
  int koff = (lane >> 4) * 8;
  const _Float16* p = A + (size_t)row * ld + k0 + koff;
  AFragU u;
  u.f[0] = *reinterpret_cast<const float4*>(p);       // K = koff .. koff+7
  u.f[1] = *reinterpret_cast<const float4*>(p + 16);  // K = 16+koff .. 16+koff+7
  return u.v;
}

// B tile 32x16 where B[k][n] = W[j0+n][k0+k], W row-major [Nc, K]
__device__ __forceinline__ v16h load_b_frag_rowT(const _Float16* __restrict__ W, int ld,
                                                 int j0, int k0, int maxRow, int lane) {
  int col = j0 + (lane & 15);
  if (col > maxRow) col = maxRow;
  int kb = k0 + (lane >> 4) * 8;
  const _Float16* p = W + (size_t)col * ld + kb;
  AFragU u;
  u.f[0] = *reinterpret_cast<const float4*>(p);      // K = kb .. kb+7
  u.f[1] = *reinterpret_cast<const float4*>(p + 16); // K = kb+16 .. kb+23
  return u.v;
}

// A frag from LDS tile As[128][64] (row-major, 16B aligned)
__device__ __forceinline__ v16h lds_a_frag(const _Float16* As, int rbase, int kk, int lane) {
  int row = rbase + (lane & 15);
  int koff = kk + (lane >> 4) * 8;
  const _Float16* p = As + row * 64 + koff;
  AFragU u;
  u.f[0] = *reinterpret_cast<const float4*>(p);
  u.f[1] = *reinterpret_cast<const float4*>(p + 16);
  return u.v;
}

// B frag from LDS tile Bs[64][64]: B[k][n] = Bs[cbase+n][kk+k]
__device__ __forceinline__ v16h lds_b_frag(const _Float16* Bs, int cbase, int kk, int lane) {
  int col = cbase + (lane & 15);
  int kb = kk + (lane >> 4) * 8;
  const _Float16* p = Bs + col * 64 + kb;
  AFragU u;
  u.f[0] = *reinterpret_cast<const float4*>(p);
  u.f[1] = *reinterpret_cast<const float4*>(p + 16);
  return u.v;
}

// B tile 32x16 where B[k][n] = V[k0+k][c0+n] (column gather, stride ld)
__device__ __forceinline__ v16h load_b_frag_col(const _Float16* __restrict__ V, int ld,
                                                int k0, int c0, int maxRow, int lane) {
  int col = c0 + (lane & 15);
  int kb = k0 + (lane >> 4) * 8;
  v16h b;
#pragma unroll
  for (int e = 0; e < 16; ++e) {
    int r = kb + e + (e & 8); if (r > maxRow) r = maxRow;
    b[e] = V[(size_t)r * ld + col];
  }
  return b;
}

// A tile 16x32 from LDS f32 scores [16][LDSW]
__device__ __forceinline__ v16h load_a_frag_lds_f32(const float* __restrict__ S, int k0, int lane) {
  int row = lane & 15;
  int koff = (lane >> 4) * 8;
  const float* p = S + row * LDSW + k0;
  v16h a;
#pragma unroll
  for (int e = 0; e < 8; ++e) a[e] = (_Float16)p[koff + e];
#pragma unroll
  for (int e = 0; e < 8; ++e) a[8 + e] = (_Float16)p[16 + koff + e];
  return a;
}

// ---------------- async double-buffered WMMA GEMM ----------------
// C[M,Nc] = A[M,K] @ W[Nc,K]^T + bias; 8 waves/block, 128x64 output tile, K staged 64-deep
#define GM_QKV     0
#define GM_OUTPROJ 1
#define GM_FC1     2
#define GM_FC2     3

__global__ __launch_bounds__(256) void gemm_wmma_kernel(
    const _Float16* __restrict__ A, int M, int K,
    const _Float16* __restrict__ W, int Nc,
    const float* __restrict__ bias,
    const float* __restrict__ resid,
    float* __restrict__ outf32,
    _Float16* __restrict__ outf16,
    _Float16* __restrict__ qo, _Float16* __restrict__ ko, _Float16* __restrict__ vo,
    int mode)
{
  __shared__ alignas(16) _Float16 Bs[2][64 * 64];   // 2 x 8 KB weight tile (shared by 8 waves)
  __shared__ alignas(16) _Float16 As[2][128 * 64];  // 2 x 16 KB activation tile

  int tid = threadIdx.x;
  int lane = tid & 31;
  int wave = tid >> 5;
  int r0blk = blockIdx.x * 128;
  int r0 = r0blk + wave * 16;
  int j0 = blockIdx.y * 64;

  // ---- staging coordinates: 256 threads x 6 async B128 = 8KB B + 16KB A per stage ----
  int bj0 = tid >> 3;               // 0..31
  int bk0 = (tid & 7) * 8;          // 8-half chunks along K
  int bj1 = bj0 + 32;               // 32..63
  const _Float16* bsrc0 = W + (size_t)(j0 + bj0) * K + bk0;
  const _Float16* bsrc1 = W + (size_t)(j0 + bj1) * K + bk0;
  int arow = tid >> 3;              // A rows handled: arow, +32, +64, +96
  int akc  = (tid & 7) * 8;
  const _Float16* asrc[4];
#pragma unroll
  for (int i = 0; i < 4; ++i) {
    int gr = min(r0blk + arow + 32 * i, M - 1);
    asrc[i] = A + (size_t)gr * K + akc;
  }

  auto stage = [&](int buf, int k0) {
    async_ld_b128(&Bs[buf][bj0 * 64 + bk0], bsrc0 + k0);
    async_ld_b128(&Bs[buf][bj1 * 64 + bk0], bsrc1 + k0);
#pragma unroll
    for (int i = 0; i < 4; ++i)
      async_ld_b128(&As[buf][(arow + 32 * i) * 64 + akc], asrc[i] + k0);
  };

  v8f acc[4] = {v8f{}, v8f{}, v8f{}, v8f{}};
  int nsteps = K >> 6;              // K staged 64-deep (K is a multiple of 64)

  stage(0, 0);
  for (int s = 0; s < nsteps; ++s) {
    int buf = s & 1;
    if (s + 1 < nsteps) {
      stage(buf ^ 1, (s + 1) << 6); // prefetch next stage while current completes
      WAIT_ASYNC(6);                // current stage's 6 ops done (in-order ASYNCcnt)
    } else {
      WAIT_ASYNC(0);
    }
    __syncthreads();
#pragma unroll
    for (int kk = 0; kk < 64; kk += 32) {
      // hoist all fragment loads so the 4 WMMAs issue back-to-back after one dscnt wait
      v16h a  = lds_a_frag(As[buf], wave * 16, kk, lane);
      v16h b0 = lds_b_frag(Bs[buf],  0, kk, lane);
      v16h b1 = lds_b_frag(Bs[buf], 16, kk, lane);
      v16h b2 = lds_b_frag(Bs[buf], 32, kk, lane);
      v16h b3 = lds_b_frag(Bs[buf], 48, kk, lane);
      acc[0] = wmma_f16(a, b0, acc[0]);
      acc[1] = wmma_f16(a, b1, acc[1]);
      acc[2] = wmma_f16(a, b2, acc[2]);
      acc[3] = wmma_f16(a, b3, acc[3]);
    }
    __syncthreads();                // all waves done reading before buf is overwritten
  }

  // ---- epilogue ----
#pragma unroll
  for (int t = 0; t < 4; ++t) {
#pragma unroll
    for (int r = 0; r < 8; ++r) {
      int row = r0 + 8 * (lane >> 4) + r;
      int col = j0 + t * 16 + (lane & 15);
      if (row >= M || col >= Nc) continue;
      float v = acc[t][r] + bias[col];
      if (mode == GM_QKV) {
        int n = row >> 4, b = row & 15;          // token m = n*16 + b
        int which = col >> 10;                    // 0=q,1=k,2=v
        int f = col & 1023;
        int h = f >> 6, hd = f & 63;
        size_t dst = ((size_t)(b * NH + h) * NTOK + n) * HD + hd;
        if (which == 0)      qo[dst] = (_Float16)(v * 0.125f);  // hd^-0.5
        else if (which == 1) ko[dst] = (_Float16)v;
        else                 vo[dst] = (_Float16)v;
      } else if (mode == GM_OUTPROJ) {
        size_t i = (size_t)row * Nc + col;
        outf32[i] = v + resid[i];
      } else if (mode == GM_FC1) {
        float gl = v / (1.f + __expf(-1.702f * v));   // quick-gelu
        outf16[(size_t)row * Nc + col] = (_Float16)gl;
      } else {
        size_t i = (size_t)row * Nc + col;
        outf32[i] = v + resid[i];
      }
    }
  }
}

// ---------------- layernorm: f32 row -> f16 row ----------------
__global__ __launch_bounds__(256) void ln_kernel(const float* __restrict__ x,
    const float* __restrict__ g, const float* __restrict__ bt,
    _Float16* __restrict__ out)
{
  int row = blockIdx.x;
  const float* xr = x + (size_t)row * DIM;
  __shared__ float smem[8];
  __shared__ float stats[2];
  int tid = threadIdx.x, lane = tid & 31, wid = tid >> 5;
  float lv[4];
  float s = 0.f;
#pragma unroll
  for (int i = 0; i < 4; ++i) { lv[i] = xr[tid + i * 256]; s += lv[i]; }
  for (int o = 16; o; o >>= 1) s += __shfl_xor(s, o, 32);
  if (lane == 0) smem[wid] = s;
  __syncthreads();
  if (tid == 0) { float t = 0; for (int i = 0; i < 8; ++i) t += smem[i]; stats[0] = t * (1.f / DIM); }
  __syncthreads();
  float mean = stats[0];
  float vv = 0.f;
#pragma unroll
  for (int i = 0; i < 4; ++i) { float d = lv[i] - mean; vv += d * d; }
  for (int o = 16; o; o >>= 1) vv += __shfl_xor(vv, o, 32);
  __syncthreads();
  if (lane == 0) smem[wid] = vv;
  __syncthreads();
  if (tid == 0) { float t = 0; for (int i = 0; i < 8; ++i) t += smem[i]; stats[1] = rsqrtf(t * (1.f / DIM) + 1e-5f); }
  __syncthreads();
  float rstd = stats[1];
#pragma unroll
  for (int i = 0; i < 4; ++i) {
    int d = tid + i * 256;
    out[(size_t)row * DIM + d] = (_Float16)((lv[i] - mean) * rstd * g[d] + bt[d]);
  }
}

// ---------------- attention: one wave per (b,h,q-strip of 16) ----------------
__global__ __launch_bounds__(32) void attn_kernel(
    const _Float16* __restrict__ Q, const _Float16* __restrict__ Km,
    const _Float16* __restrict__ V,
    _Float16* __restrict__ Aout, float* __restrict__ attnmean)
{
  __shared__ float S[16 * LDSW];
  int lane = threadIdx.x & 31;
  int q0 = blockIdx.x * 16;
  int bh = blockIdx.y;                    // b*16 + h
  int b = bh >> 4, h = bh & 15;
  const _Float16* Qp = Q + (size_t)bh * NTOK * HD;
  const _Float16* Kp = Km + (size_t)bh * NTOK * HD;
  const _Float16* Vp = V + (size_t)bh * NTOK * HD;

  for (int i = threadIdx.x; i < 16 * LDSW; i += 32) S[i] = 0.f;
  __syncthreads();

  // scores strip: 16 x 592 via WMMA over hd; Q fragments are loop-invariant
  v16h qa0 = load_a_frag(Qp, HD, q0, 0,  NTOK - 1, lane);
  v16h qa1 = load_a_frag(Qp, HD, q0, 32, NTOK - 1, lane);
  for (int kt = 0; kt < 37; ++kt) {
    v8f c = {};
    v16h b0 = load_b_frag_rowT(Kp, HD, kt * 16, 0,  NTOK - 1, lane);
    v16h b1 = load_b_frag_rowT(Kp, HD, kt * 16, 32, NTOK - 1, lane);
    c = wmma_f16(qa0, b0, c);
    c = wmma_f16(qa1, b1, c);
#pragma unroll
    for (int r = 0; r < 8; ++r) {
      int row = 8 * (lane >> 4) + r;
      int col = kt * 16 + (lane & 15);
      S[row * LDSW + col] = c[r];
    }
  }
  __syncthreads();

  // softmax per row + accumulate head-mean attention
  for (int row = 0; row < 16; ++row) {
    int qrow = q0 + row;
    float* Srow = &S[row * LDSW];
    if (qrow < NTOK) {
      float mx = -3.4e38f;
      for (int i = lane; i < NTOK; i += 32) mx = fmaxf(mx, Srow[i]);
      for (int o = 16; o; o >>= 1) mx = fmaxf(mx, __shfl_xor(mx, o, 32));
      float sum = 0.f;
      for (int i = lane; i < NTOK; i += 32) { float e = __expf(Srow[i] - mx); Srow[i] = e; sum += e; }
      for (int o = 16; o; o >>= 1) sum += __shfl_xor(sum, o, 32);
      float inv = 1.f / sum;
      for (int i = lane; i < NTOK; i += 32) {
        float p = Srow[i] * inv; Srow[i] = p;
        atomicAdd(&attnmean[((size_t)b * NTOK + qrow) * NTOK + i], p * (1.f / NH));
      }
      for (int i = NTOK + lane; i < LDSW; i += 32) Srow[i] = 0.f;
    } else {
      for (int i = lane; i < LDSW; i += 32) Srow[i] = 0.f;
    }
  }
  __syncthreads();

  // out strip = P @ V  (4 column tiles of 16)
#pragma unroll
  for (int ct = 0; ct < 4; ++ct) {
    v8f c = {};
    for (int k0 = 0; k0 < LDSW; k0 += 32) {
      v16h a = load_a_frag_lds_f32(S, k0, lane);
      v16h bb = load_b_frag_col(Vp, HD, k0, ct * 16, NTOK - 1, lane);
      c = wmma_f16(a, bb, c);
    }
#pragma unroll
    for (int r = 0; r < 8; ++r) {
      int qrow = q0 + 8 * (lane >> 4) + r;
      int hd = ct * 16 + (lane & 15);
      if (qrow < NTOK)
        Aout[((size_t)(qrow * NB + b)) * DIM + h * HD + hd] = (_Float16)c[r];
    }
  }
}

// ---------------- token scoring / selection ----------------
__global__ void col_kernel(const float* __restrict__ attnmean, float* __restrict__ col) {
  int idx = blockIdx.x * 256 + threadIdx.x;
  if (idx >= NB * NTOK) return;
  int b = idx / NTOK, k = idx % NTOK;
  float s = 0.f;
  for (int q = 0; q < NTOK; ++q) {
    float v = attnmean[((size_t)b * NTOK + q) * NTOK + k];
    float v2 = v * v; s += v2 * v2;
  }
  col[b * NTOK + k] = sqrtf(sqrtf(s));
}

__global__ __launch_bounds__(256) void colnorm_kernel(float* __restrict__ col) {
  int b = blockIdx.x;
  __shared__ float smem[8];
  __shared__ float tot;
  int tid = threadIdx.x, lane = tid & 31, wid = tid >> 5;
  float s = 0.f;
  for (int k = tid; k < NTOK; k += 256) s += col[b * NTOK + k];
  for (int o = 16; o; o >>= 1) s += __shfl_xor(s, o, 32);
  if (lane == 0) smem[wid] = s;
  __syncthreads();
  if (tid == 0) { float t = 0; for (int i = 0; i < 8; ++i) t += smem[i]; tot = 1.f / (t + 1e-6f); }
  __syncthreads();
  float inv = tot;
  for (int k = tid; k < NTOK; k += 256) col[b * NTOK + k] *= inv;
}

__global__ __launch_bounds__(576) void rank_kernel(const float* __restrict__ col,
    int* __restrict__ idx1, int* __restrict__ rankof) {
  int b = blockIdx.x, i = threadIdx.x;
  __shared__ float v[576];
  v[i] = col[b * NTOK + 1 + i];
  __syncthreads();
  float vi = v[i];
  int r = 0;
  for (int j = 0; j < 576; ++j) {
    float vj = v[j];
    if (vj > vi || (vj == vi && j < i)) ++r;
  }
  idx1[b * 576 + r] = i;     // descending order -> position r holds index i
  rankof[b * 576 + i] = r;
}

__global__ void select_kernel(const int* __restrict__ idx1, const int* __restrict__ rankof,
                              int* __restrict__ outIdx, int* __restrict__ comp) {
  int b = blockIdx.x;
  if (threadIdx.x == 0) {
    int c = 0;
    for (int i = 0; i < 576; ++i)
      if (rankof[b * 576 + i] >= LEFT) comp[b * NCOMP + c++] = i;
  }
  for (int t = threadIdx.x; t < K1; t += blockDim.x)
    outIdx[b * K1 + t] = idx1[b * 576 + t];
}

// ---------------- gather / coreset / extra ----------------
__global__ __launch_bounds__(256) void build_xo_kernel(const float* __restrict__ x2,
    const int* __restrict__ idx1, float* __restrict__ xo) {
  int row = blockIdx.x, b = blockIdx.y;
  int n = (row == 0) ? 0 : (idx1[b * 576 + row - 1] + 1);
  const float* src = x2 + ((size_t)n * NB + b) * DIM;
  float* dst = xo + ((size_t)row * NB + b) * DIM;
  for (int d = threadIdx.x; d < DIM; d += 256) dst[d] = src[d];
}

__global__ __launch_bounds__(256) void assign_kernel(const float* __restrict__ x2,
    const int* __restrict__ idx1, int* __restrict__ assign) {
  int blk = blockIdx.x;
  int b = blk / NCLUST, m = blk % NCLUST;
  const int seeds[4] = {0, 38, 76, 115};   // linspace(0,115,4).astype(int)
  int tok = idx1[b * 576 + K1 + m];
  const float* xm = x2 + ((size_t)(tok + 1) * NB + b) * DIM;
  const float* ctr[4];
#pragma unroll
  for (int c = 0; c < 4; ++c) {
    int ct = idx1[b * 576 + K1 + seeds[c]];
    ctr[c] = x2 + ((size_t)(ct + 1) * NB + b) * DIM;
  }
  float p[4] = {0.f, 0.f, 0.f, 0.f};
  for (int d = threadIdx.x; d < DIM; d += 256) {
    float xv = xm[d];
#pragma unroll
    for (int c = 0; c < 4; ++c) { float df = xv - ctr[c][d]; p[c] += df * df; }
  }
  __shared__ float acc[4][8];
  int lane = threadIdx.x & 31, wid = threadIdx.x >> 5;
#pragma unroll
  for (int c = 0; c < 4; ++c) {
    float s = p[c];
    for (int o = 16; o; o >>= 1) s += __shfl_xor(s, o, 32);
    if (lane == 0) acc[c][wid] = s;
  }
  __syncthreads();
  if (threadIdx.x == 0) {
    float best = 3.4e38f; int bi = 0;
    for (int c = 0; c < 4; ++c) {
      float t = 0; for (int w = 0; w < 8; ++w) t += acc[c][w];
      if (t < best) { best = t; bi = c; }
    }
    assign[b * NCLUST + m] = bi;
  }
}

__global__ __launch_bounds__(256) void merge_kernel(const float* __restrict__ x2,
    const int* __restrict__ idx1, const int* __restrict__ assign, float* __restrict__ xo) {
  int b = blockIdx.x >> 2, c = blockIdx.x & 3;
  __shared__ int toks[NCLUST];
  __shared__ int cnt_s;
  if (threadIdx.x == 0) {
    int cnt = 0;
    for (int m = 0; m < NCLUST; ++m)
      if (assign[b * NCLUST + m] == c) toks[cnt++] = idx1[b * 576 + K1 + m];
    cnt_s = cnt;
  }
  __syncthreads();
  int cnt = cnt_s;
  float inv = 1.f / fmaxf((float)cnt, 1.f);
  for (int d = threadIdx.x; d < DIM; d += 256) {
    float s = 0.f;
    for (int m = 0; m < cnt; ++m) s += x2[((size_t)(toks[m] + 1) * NB + b) * DIM + d];
    xo[((size_t)(1 + K1 + c) * NB + b) * DIM + d] = s * inv;
  }
}

__global__ __launch_bounds__(256) void extra_kernel(const float* __restrict__ x2,
    const int* __restrict__ comp, const float* __restrict__ col, float* __restrict__ xo) {
  int b = blockIdx.x;
  __shared__ float w[NCOMP];
  __shared__ int ci[NCOMP];
  if (threadIdx.x < NCOMP) {
    int i = comp[b * NCOMP + threadIdx.x];
    ci[threadIdx.x] = i;
    w[threadIdx.x] = col[b * NTOK + 1 + i];
  }
  __syncthreads();
  for (int d = threadIdx.x; d < DIM; d += 256) {
    float s = 0.f;
    for (int t = 0; t < NCOMP; ++t)
      s += x2[((size_t)(ci[t] + 1) * NB + b) * DIM + d] * w[t];
    xo[((size_t)(NOUT - 1) * NB + b) * DIM + d] = s;
  }
}

// ---------------- f32 -> f16 conversion ----------------
__global__ void cvt_f16_kernel(const float* __restrict__ src, _Float16* __restrict__ dst, size_t n) {
  size_t i = blockIdx.x * (size_t)blockDim.x + threadIdx.x;
  if (i < n) dst[i] = (_Float16)src[i];
}

// ---------------- launch ----------------
extern "C" void kernel_launch(void* const* d_in, const int* in_sizes, int n_in,
                              void* d_out, int out_size, void* d_ws, size_t ws_size,
                              hipStream_t stream) {
  const float* x         = (const float*)d_in[0];
  const float* in_proj_w = (const float*)d_in[1];
  const float* in_proj_b = (const float*)d_in[2];
  const float* out_proj_w= (const float*)d_in[3];
  const float* out_proj_b= (const float*)d_in[4];
  const float* ln1_g     = (const float*)d_in[5];
  const float* ln1_b     = (const float*)d_in[6];
  const float* ln2_g     = (const float*)d_in[7];
  const float* ln2_b     = (const float*)d_in[8];
  const float* fc1_w     = (const float*)d_in[9];
  const float* fc1_b     = (const float*)d_in[10];
  const float* fc2_w     = (const float*)d_in[11];
  const float* fc2_b     = (const float*)d_in[12];

  char* ws = (char*)d_ws;
  size_t off = 0;
  auto alloc = [&](size_t bytes) { size_t r = off; off = (off + bytes + 255) & ~(size_t)255; return r; };

  _Float16* w_in_h  = (_Float16*)(ws + alloc((size_t)3 * DIM * DIM * 2));
  _Float16* w_out_h = (_Float16*)(ws + alloc((size_t)DIM * DIM * 2));
  _Float16* w_fc1_h = (_Float16*)(ws + alloc((size_t)FF * DIM * 2));
  _Float16* w_fc2_h = (_Float16*)(ws + alloc((size_t)DIM * FF * 2));
  _Float16* h1      = (_Float16*)(ws + alloc((size_t)MTOK * DIM * 2));
  _Float16* qf      = (_Float16*)(ws + alloc((size_t)NB * NH * NTOK * HD * 2));
  _Float16* kf      = (_Float16*)(ws + alloc((size_t)NB * NH * NTOK * HD * 2));
  _Float16* vf      = (_Float16*)(ws + alloc((size_t)NB * NH * NTOK * HD * 2));
  _Float16* aout    = (_Float16*)(ws + alloc((size_t)MTOK * DIM * 2));
  float*    attnmean= (float*)   (ws + alloc((size_t)NB * NTOK * NTOK * 4));
  float*    x2      = (float*)   (ws + alloc((size_t)MTOK * DIM * 4));
  float*    col     = (float*)   (ws + alloc((size_t)NB * NTOK * 4));
  int*      idx1    = (int*)     (ws + alloc((size_t)NB * 576 * 4));
  int*      rankof  = (int*)     (ws + alloc((size_t)NB * 576 * 4));
  int*      comp    = (int*)     (ws + alloc((size_t)NB * NCOMP * 4));
  int*      assign  = (int*)     (ws + alloc((size_t)NB * NCLUST * 4));
  float*    xo      = (float*)   (ws + alloc((size_t)MOUT * DIM * 4));
  _Float16* h2      = (_Float16*)(ws + alloc((size_t)MOUT * DIM * 2));
  _Float16* mid     = (_Float16*)(ws + alloc((size_t)MOUT * FF * 2));

  float* outF  = (float*)d_out;                       // (409,16,1024)
  int*   outIdx= (int*)d_out + (size_t)MOUT * DIM;    // (16,403)

  // weight conversions
  {
    size_t n0 = (size_t)3 * DIM * DIM, n1 = (size_t)DIM * DIM,
           n2 = (size_t)FF * DIM,      n3 = (size_t)DIM * FF;
    cvt_f16_kernel<<<(unsigned)((n0 + 255) / 256), 256, 0, stream>>>(in_proj_w, w_in_h, n0);
    cvt_f16_kernel<<<(unsigned)((n1 + 255) / 256), 256, 0, stream>>>(out_proj_w, w_out_h, n1);
    cvt_f16_kernel<<<(unsigned)((n2 + 255) / 256), 256, 0, stream>>>(fc1_w, w_fc1_h, n2);
    cvt_f16_kernel<<<(unsigned)((n3 + 255) / 256), 256, 0, stream>>>(fc2_w, w_fc2_h, n3);
  }

  // LN1
  ln_kernel<<<MTOK, 256, 0, stream>>>(x, ln1_g, ln1_b, h1);

  // zero attn head-mean accumulator
  hipMemsetAsync(attnmean, 0, (size_t)NB * NTOK * NTOK * 4, stream);

  // QKV projection  (577 row-tiles -> 73 blocks of 8 waves)
  gemm_wmma_kernel<<<dim3((MTOK / 16 + 7) / 8, (3 * DIM) / 64), 256, 0, stream>>>(
      h1, MTOK, DIM, w_in_h, 3 * DIM, in_proj_b,
      nullptr, nullptr, nullptr, qf, kf, vf, GM_QKV);

  // attention
  attn_kernel<<<dim3(37, NB * NH), 32, 0, stream>>>(qf, kf, vf, aout, attnmean);

  // out projection + residual -> x2
  gemm_wmma_kernel<<<dim3((MTOK / 16 + 7) / 8, DIM / 64), 256, 0, stream>>>(
      aout, MTOK, DIM, w_out_h, DIM, out_proj_b,
      x, x2, nullptr, nullptr, nullptr, nullptr, GM_OUTPROJ);

  // token scoring
  col_kernel<<<(NB * NTOK + 255) / 256, 256, 0, stream>>>(attnmean, col);
  colnorm_kernel<<<NB, 256, 0, stream>>>(col);
  rank_kernel<<<NB, 576, 0, stream>>>(col, idx1, rankof);
  select_kernel<<<NB, 256, 0, stream>>>(idx1, rankof, outIdx, comp);

  // build output token set
  build_xo_kernel<<<dim3(1 + K1, NB), 256, 0, stream>>>(x2, idx1, xo);
  assign_kernel<<<NB * NCLUST, 256, 0, stream>>>(x2, idx1, assign);
  merge_kernel<<<NB * 4, 256, 0, stream>>>(x2, idx1, assign, xo);
  extra_kernel<<<NB, 256, 0, stream>>>(x2, comp, col, xo);

  // LN2 + MLP
  ln_kernel<<<MOUT, 256, 0, stream>>>(xo, ln2_g, ln2_b, h2);
  gemm_wmma_kernel<<<dim3((MOUT / 16 + 7) / 8, FF / 64), 256, 0, stream>>>(
      h2, MOUT, DIM, w_fc1_h, FF, fc1_b,
      nullptr, nullptr, mid, nullptr, nullptr, nullptr, GM_FC1);
  gemm_wmma_kernel<<<dim3((MOUT / 16 + 7) / 8, DIM / 64), 256, 0, stream>>>(
      mid, MOUT, FF, w_fc2_h, DIM, fc2_b,
      xo, outF, nullptr, nullptr, nullptr, nullptr, GM_FC2);
}
